// RadixAttentionBlock_41583873360700
// MI455X (gfx1250) — matrix operand order repaired and checked by reading, not compiled, not run on hardware
//
#include <hip/hip_runtime.h>
#include <hip/hip_bf16.h>

// ---------------------------------------------------------------------------
// Types for CDNA5 WMMA (wave32, v_wmma_f32_16x16x32_bf16)
// ---------------------------------------------------------------------------
typedef __bf16 bf16;
typedef __attribute__((ext_vector_type(16))) __bf16 v16bf;
typedef __attribute__((ext_vector_type(8)))  __bf16 v8bf;
typedef __attribute__((ext_vector_type(4)))  __bf16 v4bf;
typedef __attribute__((ext_vector_type(2)))  __bf16 v2bf;
typedef __attribute__((ext_vector_type(8)))  float  v8f;

__device__ __forceinline__ v8f wmma_bf16(v16bf a, v16bf b, v8f c) {
  // (neg_a, A, neg_b, B, c_mod, C, reuse_a, reuse_b)
  return __builtin_amdgcn_wmma_f32_16x16x32_bf16(false, a, false, b, (short)0, c,
                                                 false, false);
}

// Concatenate two 8x-bf16 (16B) LDS chunks into one 16-element fragment.
__device__ __forceinline__ v16bf frag_cat(const bf16* p0, const bf16* p1) {
  v8bf lo = *(const v8bf*)p0;
  v8bf hi = *(const v8bf*)p1;
  v16bf r;
#pragma unroll
  for (int i = 0; i < 8; ++i) { r[i] = lo[i]; r[i + 8] = hi[i]; }
  return r;
}

// A-fragment (16x32, MxK): lane L holds row rowBase + (L&15);
// element idx -> K = kOff + (idx>=8)*16 + (L>=16)*8 + (idx&7)
__device__ __forceinline__ v16bf load_afrag(const bf16* tile, int rowBase, int rs,
                                            int kOff) {
  const int lane = threadIdx.x & 31;
  const int hi = lane >> 4;
  const bf16* p = tile + (rowBase + (lane & 15)) * rs + kOff + hi * 8;
  return frag_cat(p, p + 16);
}

// B-fragment (32x16, KxN) built from an LDS tile stored as [n][k]:
// lane L holds column colBase + (L&15); element idx -> K = kOff + (idx>=8)*16 + (L>=16)*8 + (idx&7)
__device__ __forceinline__ v16bf load_bfrag(const bf16* tile, int colBase, int rs,
                                            int kOff) {
  const int lane = threadIdx.x & 31;
  const int hi = lane >> 4;
  const bf16* p = tile + (colBase + (lane & 15)) * rs + kOff + hi * 8;
  return frag_cat(p, p + 16);
}

// CDNA5 async DMA: global -> LDS, 16B per lane, tracked by ASYNCcnt.
__device__ __forceinline__ void async_copy_b128(const bf16* gsrc, bf16* lds_dst) {
  unsigned off = (unsigned)(uintptr_t)lds_dst;  // low 32 bits = LDS byte offset
  asm volatile("global_load_async_to_lds_b128 %0, %1, off"
               :: "v"(off), "v"(gsrc)
               : "memory");
}
__device__ __forceinline__ void wait_async() {
  asm volatile("s_wait_asynccnt 0x0" ::: "memory");
}

// ---------------------------------------------------------------------------
// C[4096x2048] = A[4096x2048] (f32) x W[2048x2048]^T (f32, torch Linear layout)
// 64x128 tile per 128-thread block (4 waves x 16 rows x 8 N-subtiles).
// K staged in 32-wide bf16 LDS tiles, double-buffered: global loads for tile
// k+1 issue before the WMMAs of tile k, hiding HBM latency behind matrix math.
// ---------------------------------------------------------------------------
template <bool BF16OUT>
__global__ __launch_bounds__(128) void gemm_nt(const float* __restrict__ A,
                                               const float* __restrict__ W,
                                               void* __restrict__ outp) {
  constexpr int RS = 40;  // 32 + 8 halfs pad: 80B rows, 16B aligned, conflict-free
  __shared__ bf16 As[2][64 * RS];
  __shared__ bf16 Bs[2][128 * RS];

  const int tid = threadIdx.x;
  const int lane = tid & 31, wid = tid >> 5, hi = lane >> 4;
  const int m0 = blockIdx.y * 64, n0 = blockIdx.x * 128;

  // Staging assignment: A rows 64x32 (half row per thread), B rows 128x32
  // (full 32-wide row per thread).
  const int ar = tid >> 1, ac = (tid & 1) * 16;
  const float* Aptr = A + (size_t)(m0 + ar) * 2048 + ac;
  const float* Wptr = W + (size_t)(n0 + tid) * 2048;

  float4 ra[4], rb[8];
  auto preload = [&](int kb) {
#pragma unroll
    for (int i = 0; i < 4; ++i) ra[i] = *(const float4*)(Aptr + kb + i * 4);
#pragma unroll
    for (int i = 0; i < 8; ++i) rb[i] = *(const float4*)(Wptr + kb + i * 4);
  };
  auto stage = [&](int buf) {
    bf16* da = &As[buf][ar * RS + ac];
#pragma unroll
    for (int i = 0; i < 4; ++i) {
      da[i * 4 + 0] = (bf16)ra[i].x; da[i * 4 + 1] = (bf16)ra[i].y;
      da[i * 4 + 2] = (bf16)ra[i].z; da[i * 4 + 3] = (bf16)ra[i].w;
    }
    bf16* db = &Bs[buf][tid * RS];
#pragma unroll
    for (int i = 0; i < 8; ++i) {
      db[i * 4 + 0] = (bf16)rb[i].x; db[i * 4 + 1] = (bf16)rb[i].y;
      db[i * 4 + 2] = (bf16)rb[i].z; db[i * 4 + 3] = (bf16)rb[i].w;
    }
  };

  v8f acc[8];
#pragma unroll
  for (int t = 0; t < 8; ++t) acc[t] = (v8f){0.f, 0.f, 0.f, 0.f, 0.f, 0.f, 0.f, 0.f};

  preload(0);
  stage(0);
  __syncthreads();

  for (int kb = 0, it = 0; kb < 2048; kb += 32, ++it) {
    const int buf = it & 1;
    const bool more = (kb + 32) < 2048;
    if (more) preload(kb + 32);  // global loads issue before compute (double buffer)
    if (kb + 64 < 2048) {        // prefetch two tiles ahead into L2
      __builtin_prefetch(Aptr + kb + 64, 0, 1);
      __builtin_prefetch(Wptr + kb + 64, 0, 1);
    }

    v16bf af = load_afrag(As[buf], wid * 16, RS, 0);
#pragma unroll
    for (int g = 0; g < 2; ++g) {
      v16bf bt[4];
#pragma unroll
      for (int t = 0; t < 4; ++t) bt[t] = load_bfrag(Bs[buf], g * 64 + t * 16, RS, 0);
#pragma unroll
      for (int t = 0; t < 4; ++t) acc[g * 4 + t] = wmma_bf16(af, bt[t], acc[g * 4 + t]);
    }

    if (more) stage(buf ^ 1);
    __syncthreads();
  }

#pragma unroll
  for (int t = 0; t < 8; ++t)
#pragma unroll
    for (int r = 0; r < 8; ++r) {
      const int row = m0 + wid * 16 + r + hi * 8;
      const int col = n0 + t * 16 + (lane & 15);
      const float v = acc[t][r];
      if constexpr (BF16OUT)
        ((bf16*)outp)[(size_t)row * 2048 + col] = (bf16)v;
      else
        ((float*)outp)[(size_t)row * 2048 + col] = v;
    }
}

// ---------------------------------------------------------------------------
// In-place RoPE on bf16 Q and K, [4096 tokens][16 heads x 128].
// Each thread owns rotary indices (2i, 2i+1) and their partners (+64), so all
// global traffic is 32-bit. Folds 1/sqrt(128) into Q.
// ---------------------------------------------------------------------------
__global__ __launch_bounds__(256) void rope_inplace(bf16* __restrict__ Qb,
                                                    bf16* __restrict__ Kb) {
  const int gid = blockIdx.x * 256 + threadIdx.x;  // 4096*32 threads
  const int tok = gid >> 5;
  const int i = (gid & 31) * 2;  // rotary pair base: handles i and i+1
  const float pos = (float)(tok & 2047);
  const float a0 = pos * __powf(10000.f, -(float)i * (1.0f / 64.0f));
  const float a1 = pos * __powf(10000.f, -(float)(i + 1) * (1.0f / 64.0f));
  const float c0 = __cosf(a0), s0 = __sinf(a0);
  const float c1 = __cosf(a1), s1 = __sinf(a1);
  const float qs = 0.088388347648318447f;  // 1/sqrt(128)
  const size_t rb = (size_t)tok * 2048 + i;
#pragma unroll
  for (int h = 0; h < 16; ++h) {
    const size_t a = rb + h * 128;
    {
      v2bf x1 = *(v2bf*)&Qb[a], x2 = *(v2bf*)&Qb[a + 64];
      const float q10 = (float)x1[0], q11 = (float)x1[1];
      const float q20 = (float)x2[0], q21 = (float)x2[1];
      v2bf y1, y2;
      y1[0] = (bf16)((q10 * c0 - q20 * s0) * qs);
      y1[1] = (bf16)((q11 * c1 - q21 * s1) * qs);
      y2[0] = (bf16)((q20 * c0 + q10 * s0) * qs);
      y2[1] = (bf16)((q21 * c1 + q11 * s1) * qs);
      *(v2bf*)&Qb[a] = y1;
      *(v2bf*)&Qb[a + 64] = y2;
    }
    {
      v2bf x1 = *(v2bf*)&Kb[a], x2 = *(v2bf*)&Kb[a + 64];
      const float k10 = (float)x1[0], k11 = (float)x1[1];
      const float k20 = (float)x2[0], k21 = (float)x2[1];
      v2bf y1, y2;
      y1[0] = (bf16)(k10 * c0 - k20 * s0);
      y1[1] = (bf16)(k11 * c1 - k21 * s1);
      y2[0] = (bf16)(k20 * c0 + k10 * s0);
      y2[1] = (bf16)(k21 * c1 + k11 * s1);
      *(v2bf*)&Kb[a] = y1;
      *(v2bf*)&Kb[a + 64] = y2;
    }
  }
}

// ---------------------------------------------------------------------------
// Flash-style attention, one (b, h, 64-row q-tile) per 128-thread block.
// Q/K/V bf16 stored as [token][h*128+d]. Output f32 same layout.
// Q and K tiles staged with CDNA5 async global->LDS DMA (ASYNCcnt); V staged
// through a register transpose: each thread owns 4 keys x 16 d, packing 4
// keys per d-row into 8-byte LDS stores (b64) instead of scalar b16s.
// ---------------------------------------------------------------------------
__global__ __launch_bounds__(128) void attn(const bf16* __restrict__ Q,
                                            const bf16* __restrict__ K,
                                            const bf16* __restrict__ V,
                                            float* __restrict__ O) {
  constexpr int QRS = 136;  // 128 + 8 pad (272B rows, 16B aligned)
  constexpr int PRS = 72;   // 64 + 8 pad (144B rows, 16B & 8B aligned)
  __shared__ bf16 Qs[64 * QRS];     // 17408 B
  __shared__ bf16 Ks[64 * QRS];     // 17408 B
  __shared__ bf16 VsT[128 * PRS];   // 18432 B, transposed: [d][key]
  __shared__ bf16 Ps[4 * 16 * PRS]; // 9216 B, per-wave P tiles

  const int tid = threadIdx.x, lane = tid & 31, wid = tid >> 5, hi = lane >> 4;
  const int bh = blockIdx.y;  // b*16 + h
  const int b = bh >> 4, h = bh & 15;
  const int q0 = blockIdx.x * 64;
  const size_t base = (size_t)b * 2048 * 2048 + (size_t)h * 128;

  const int sr = tid >> 1, scb = (tid & 1) * 64;  // async staging: row, col-base
  const int vk = (tid & 15) * 4;                  // V staging: 4 keys
  const int vd = (tid >> 4) * 16;                 // V staging: 16 d-values

  // Stage Q tile (64 x 128 bf16) via async DMA; already RoPE'd and pre-scaled.
  {
    const bf16* src = Q + base + (size_t)(q0 + sr) * 2048 + scb;
    bf16* dst = &Qs[sr * QRS + scb];
#pragma unroll
    for (int i = 0; i < 8; ++i) async_copy_b128(src + i * 8, dst + i * 8);
  }
  wait_async();
  __syncthreads();

  v16bf aq[4];
#pragma unroll
  for (int kk = 0; kk < 4; ++kk) aq[kk] = load_afrag(Qs, wid * 16, QRS, kk * 32);

  v8f o[8];
#pragma unroll
  for (int t = 0; t < 8; ++t) o[t] = (v8f){0.f, 0.f, 0.f, 0.f, 0.f, 0.f, 0.f, 0.f};
  float m_i[8], l_i[8];
#pragma unroll
  for (int r = 0; r < 8; ++r) { m_i[r] = -1e30f; l_i[r] = 0.f; }

  bf16* Psw = &Ps[wid * 16 * PRS];

  for (int kc = 0; kc < 2048; kc += 64) {
    // Async-stage K chunk row-major; overlap with V register-transpose staging.
    {
      const bf16* sk = K + base + (size_t)(kc + sr) * 2048 + scb;
      bf16* dk = &Ks[sr * QRS + scb];
#pragma unroll
      for (int i = 0; i < 8; ++i) async_copy_b128(sk + i * 8, dk + i * 8);

      // V: read 4 key-rows x 16 d, transpose in registers, store packed b64.
      v8bf rv[4][2];
#pragma unroll
      for (int k = 0; k < 4; ++k)
#pragma unroll
        for (int c = 0; c < 2; ++c)
          rv[k][c] = *(const v8bf*)(V + base + (size_t)(kc + vk + k) * 2048 + vd +
                                    c * 8);
#pragma unroll
      for (int c = 0; c < 2; ++c)
#pragma unroll
        for (int j = 0; j < 8; ++j) {
          v4bf w;
          w[0] = rv[0][c][j]; w[1] = rv[1][c][j];
          w[2] = rv[2][c][j]; w[3] = rv[3][c][j];
          *(v4bf*)&VsT[(vd + c * 8 + j) * PRS + vk] = w;
        }
    }
    wait_async();
    __syncthreads();

    // scores S = Q K^T : 16 q-rows x 64 keys per wave
    v8f sc[4];
#pragma unroll
    for (int t = 0; t < 4; ++t) {
      v16bf bk[4];
#pragma unroll
      for (int kk = 0; kk < 4; ++kk) bk[kk] = load_bfrag(Ks, t * 16, QRS, kk * 32);
      sc[t] = (v8f){0.f, 0.f, 0.f, 0.f, 0.f, 0.f, 0.f, 0.f};
#pragma unroll
      for (int kk = 0; kk < 4; ++kk) sc[t] = wmma_bf16(aq[kk], bk[kk], sc[t]);
    }

    // Online softmax per row (row = r + hi*8 lives in one 16-lane half).
#pragma unroll
    for (int r = 0; r < 8; ++r) {
      float mx = fmaxf(fmaxf(sc[0][r], sc[1][r]), fmaxf(sc[2][r], sc[3][r]));
#pragma unroll
      for (int off = 1; off < 16; off <<= 1) mx = fmaxf(mx, __shfl_xor(mx, off, 32));
      const float mnew = fmaxf(m_i[r], mx);
      const float rescale = __expf(m_i[r] - mnew);
      m_i[r] = mnew;
      l_i[r] *= rescale;
#pragma unroll
      for (int t2 = 0; t2 < 8; ++t2) o[t2][r] *= rescale;
      float psum = 0.f;
#pragma unroll
      for (int t = 0; t < 4; ++t) {
        const float p = __expf(sc[t][r] - mnew);
        psum += p;
        Psw[(r + hi * 8) * PRS + t * 16 + (lane & 15)] = (bf16)p;
      }
      l_i[r] += psum;  // per-lane partial; reduced across lanes at the end
    }

    // O += P V : contraction over the 64 keys (two 32-wide k-steps)
#pragma unroll
    for (int kk2 = 0; kk2 < 2; ++kk2) {
      v16bf ap = load_afrag(Psw, 0, PRS, kk2 * 32);
#pragma unroll
      for (int g = 0; g < 2; ++g) {
        v16bf bv[4];
#pragma unroll
        for (int t = 0; t < 4; ++t)
          bv[t] = load_bfrag(VsT, (g * 4 + t) * 16, PRS, kk2 * 32);
#pragma unroll
        for (int t = 0; t < 4; ++t)
          o[g * 4 + t] = wmma_bf16(ap, bv[t], o[g * 4 + t]);
      }
    }
    __syncthreads();
  }

  // Normalize and write out (f32, [token][h*128+d]).
#pragma unroll
  for (int r = 0; r < 8; ++r) {
    float l = l_i[r];
#pragma unroll
    for (int off = 1; off < 16; off <<= 1) l += __shfl_xor(l, off, 32);
    const float inv = 1.f / l;
    const int row = q0 + wid * 16 + r + hi * 8;
#pragma unroll
    for (int t2 = 0; t2 < 8; ++t2)
      O[base + (size_t)row * 2048 + t2 * 16 + (lane & 15)] = o[t2][r] * inv;
  }
}

// ---------------------------------------------------------------------------
// Host-side launcher. Inputs: hidden_states, wq, wk, wv, wo (all f32).
// Workspace layout (80 MB): Qb/Kb/Vb bf16 [4096][2048], Of f32 [4096][2048].
// ---------------------------------------------------------------------------
extern "C" void kernel_launch(void* const* d_in, const int* in_sizes, int n_in,
                              void* d_out, int out_size, void* d_ws, size_t ws_size,
                              hipStream_t stream) {
  (void)in_sizes; (void)n_in; (void)out_size; (void)ws_size;
  const float* X  = (const float*)d_in[0];
  const float* wq = (const float*)d_in[1];
  const float* wk = (const float*)d_in[2];
  const float* wv = (const float*)d_in[3];
  const float* wo = (const float*)d_in[4];
  float* out = (float*)d_out;

  char* ws = (char*)d_ws;
  bf16*  Qb = (bf16*)(ws);
  bf16*  Kb = (bf16*)(ws + (size_t)16 * 1024 * 1024);
  bf16*  Vb = (bf16*)(ws + (size_t)32 * 1024 * 1024);
  float* Of = (float*)(ws + (size_t)48 * 1024 * 1024);

  const dim3 gGemm(16, 64), bGemm(128);  // 128-wide N tiles, 64-wide M tiles
  // Q/K/V projections (bf16 outputs, f32 accumulation inside WMMA)
  gemm_nt<true><<<gGemm, bGemm, 0, stream>>>(X, wq, Qb);
  gemm_nt<true><<<gGemm, bGemm, 0, stream>>>(X, wk, Kb);
  gemm_nt<true><<<gGemm, bGemm, 0, stream>>>(X, wv, Vb);
  // RoPE (in place, folds 1/sqrt(hd) into Q)
  rope_inplace<<<512, 256, 0, stream>>>(Qb, Kb);
  // Attention: grid = (q-tiles, b*heads)
  attn<<<dim3(32, 32), 128, 0, stream>>>(Qb, Kb, Vb, Of);
  // Output projection (f32 out)
  gemm_nt<false><<<gGemm, bGemm, 0, stream>>>(Of, wo, out);
}
